// Network_19834158973589
// MI455X (gfx1250) — matrix operand order repaired and checked by reading, not compiled, not run on hardware
//
#include <hip/hip_runtime.h>
#include <cstdint>

// ---------------------------------------------------------------------------
// Sizes (from the reference)
// ---------------------------------------------------------------------------
#define T_STEPS 256      // 2 * MAX_LEN
#define BATCH   1024
#define EMB     300
#define EMBP    320      // emb K padded to 10 chunks of 32 (bf16 copy)
#define VOCAB   50000
#define HID     50
#define NG      200      // 4*H gate width
#define NGP     208      // Gx column stride (13 tiles of 16)
#define KIN     300
#define KH      64       // recurrent K padded 50 -> 64 (2 x 32)
#define BN_SCALE 0.999500374625937f   // 1/sqrt(1.001)

#define NTILEP  16       // N-tiles padded 13 -> 16: 4 per wave, zero guards
#define NP      (NTILEP * 16)                       // 256
#define KCHUNKS 10
#define WFRAG_ELEMS (NTILEP * KCHUNKS * 32 * 16)    // B-fragment-ordered weights

typedef __attribute__((ext_vector_type(16))) __bf16 v16bf;
typedef __attribute__((ext_vector_type(8)))  __bf16 v8bf;
typedef __attribute__((ext_vector_type(8)))  float  v8f;

__device__ __forceinline__ v16bf cat8(v8bf a, v8bf b) {
  return __builtin_shufflevector(a, b, 0,1,2,3,4,5,6,7,8,9,10,11,12,13,14,15);
}
__device__ __forceinline__ float sigm(float x) { return 1.0f / (1.0f + __expf(-x)); }

// gfx1250 async global->LDS copy (ASYNCcnt-tracked), CDNA5 ISA async ops
__device__ __forceinline__ void async_ld_b128(uint32_t lds_addr, const void* gaddr) {
  asm volatile("global_load_async_to_lds_b128 %0, %1, off"
               :: "v"(lds_addr), "v"(gaddr) : "memory");
}
__device__ __forceinline__ void wait_async0() {
  asm volatile("s_wait_asynccnt 0" ::: "memory");
}

// ---------------------------------------------------------------------------
// Pre-pass A: embedding f32 [50000][300] -> bf16 [50000][320] (zero-padded K)
// ---------------------------------------------------------------------------
__global__ __launch_bounds__(256)
void conv_emb_kernel(const float* __restrict__ emb, __bf16* __restrict__ out)
{
  const int idx = blockIdx.x * 256 + threadIdx.x;         // exactly VOCAB*EMBP
  const int v = idx / EMBP, k = idx - v * EMBP;
  out[idx] = (k < EMB) ? (__bf16)emb[(size_t)v * EMB + k] : (__bf16)0.0f;
}

// ---------------------------------------------------------------------------
// Pre-pass B: W[350][200] rows 0..299 -> bf16 in exact WMMA B-fragment order,
// padded to 16 N-tiles so the consumer has NO guards:
//   frag(nt,kc): lane l, elem e -> k = kc*32 + (l>>4)*16 + e, n = nt*16 + (l&15)
//   linear index = (((nt*KCHUNKS)+kc)*32 + l)*16 + e
// ---------------------------------------------------------------------------
__global__ __launch_bounds__(256)
void conv_wfrag_kernel(const float* __restrict__ W, __bf16* __restrict__ out)
{
  const int idx = blockIdx.x * 256 + threadIdx.x;         // exactly WFRAG_ELEMS
  const int e  = idx & 15;
  const int l  = (idx >> 4) & 31;
  const int r  = idx >> 9;
  const int kc = r % KCHUNKS;
  const int nt = r / KCHUNKS;
  const int k  = kc * 32 + (l >> 4) * 16 + e;
  const int n  = nt * 16 + (l & 15);
  out[idx] = (k < KIN && n < NG) ? (__bf16)W[(size_t)k * NG + n] : (__bf16)0.0f;
}

// ---------------------------------------------------------------------------
// Kernel 1: fused embedding gather + X @ W + bias -> bf16 gates.
// Wave w owns N-tiles 4w..4w+3 (guard-free, straight-line WMMA chains).
// grid = (numT, BATCH/16), block = 128 (4 waves).
// ---------------------------------------------------------------------------
__global__ __launch_bounds__(128)
void gates_kernel(const int* __restrict__ x1, const int* __restrict__ x2,
                  const __bf16* __restrict__ embbf, const __bf16* __restrict__ wfrag,
                  const float* __restrict__ bias, __bf16* __restrict__ outp,
                  int t_base)
{
  const int tl   = blockIdx.x;
  const int tc   = t_base + tl;
  const int b0   = blockIdx.y * 16;
  const int lane = threadIdx.x & 31;
  const int wave = threadIdx.x >> 5;
  const int m16  = lane & 15;
  const int hi   = lane >> 4;

  const int brow  = b0 + m16;
  const int token = (tc < 128) ? x1[brow * 128 + tc] : x2[brow * 128 + (tc - 128)];
  const __bf16* er = embbf + (size_t)token * EMBP;

  #pragma unroll
  for (int i = 0; i < 4; ++i) {
    const int nt = wave * 4 + i;                          // 0..15, no guard
    const __bf16* wf = wfrag + ((size_t)(nt * KCHUNKS) * 32 + lane) * 16;
    v8f acc = {};
    for (int kc = 0; kc < KCHUNKS; ++kc) {
      const int ka = kc * 32 + hi * 16;
      v8bf a0 = *(const v8bf*)&er[ka];
      v8bf a1 = *(const v8bf*)&er[ka + 8];
      const __bf16* wk = wf + (size_t)kc * (32 * 16);
      v8bf q0 = *(const v8bf*)&wk[0];
      v8bf q1 = *(const v8bf*)&wk[8];
      acc = __builtin_amdgcn_wmma_f32_16x16x32_bf16(false, cat8(a0, a1),
                                                    false, cat8(q0, q1),
                                                    (short)0, acc, false, false);
    }
    // D layout: lane l -> col n=l&15, VGPR r -> row m = r + 8*(l>>4)
    const int n = nt * 16 + m16;
    if (n < NG) {                                         // store-only guard
      const float bb = bias[n];
      #pragma unroll
      for (int r = 0; r < 8; ++r) {
        const int m = r + 8 * hi;
        outp[((size_t)tl * BATCH + b0 + m) * NGP + n] = (__bf16)(acc[r] + bb);
      }
    }
  }
}

// ---------------------------------------------------------------------------
// Kernel 2: forward LSTM scan. One WG per 16-row batch tile. W_h^T (padded to
// 256 cols) + h + c in LDS; Gx double-buffered via async global->LDS overlapped
// with the current step's WMMA + cell update. Guard-free WMMA phase.
// ---------------------------------------------------------------------------
__global__ __launch_bounds__(128)
void lstm_scan_kernel(const __bf16* __restrict__ gx, const float* __restrict__ W,
                      float* __restrict__ hfw)
{
  __shared__ __align__(16) __bf16 sWh[NP * KH];      // Wh^T [n][k], zero-padded (32 KB)
  __shared__ __align__(16) __bf16 sH [16 * KH];      // h tile (bf16, K padded)
  __shared__ __align__(16) __bf16 sGx[2][16 * NGP];  // async-staged gate tiles
  __shared__ float sC[16 * HID];
  __shared__ float sG[16 * NP];                      // gate accum (16 KB)

  const int tid  = threadIdx.x;
  const int lane = tid & 31;
  const int wave = tid >> 5;
  const int m16  = lane & 15;
  const int hi   = lane >> 4;
  const int b0   = blockIdx.x * 16;

  for (int idx = tid; idx < NP * KH; idx += 128) {
    const int n = idx >> 6, k = idx & 63;
    sWh[idx] = (k < HID && n < NG) ? (__bf16)W[(size_t)(EMB + k) * NG + n] : (__bf16)0.0f;
  }
  for (int idx = tid; idx < 16 * KH;  idx += 128) sH[idx] = (__bf16)0.0f;
  for (int idx = tid; idx < 16 * HID; idx += 128) sC[idx] = 0.0f;

  // stage Gx tile for t into sGx[buf]: 16 rows x 416 B = 26 b128 per row
  auto issue_tile = [&](int t, int buf) {
    const char* base = (const char*)(gx + ((size_t)t * BATCH + b0) * NGP);
    for (int i = tid; i < 16 * 26; i += 128) {
      const int row = i / 26, c = i - row * 26;
      const void* g = base + (size_t)row * (NGP * 2) + c * 16;
      const uint32_t l = (uint32_t)(uintptr_t)&sGx[buf][row * NGP + c * 8];
      async_ld_b128(l, g);
    }
  };

  issue_tile(0, 0);
  wait_async0();
  __syncthreads();

  for (int t = 0; t < T_STEPS; ++t) {
    if (t + 1 < T_STEPS) issue_tile(t + 1, (t + 1) & 1);   // overlap next fetch

    // ---- recurrent GEMM: [16 x 64] @ [64 x 256] -- straight-line, no guards
    v16bf Af[2];
    #pragma unroll
    for (int kc = 0; kc < 2; ++kc) {
      const int ka = kc * 32 + hi * 16;
      v8bf a0 = *(const v8bf*)&sH[m16 * KH + ka];
      v8bf a1 = *(const v8bf*)&sH[m16 * KH + ka + 8];
      Af[kc] = cat8(a0, a1);
    }
    #pragma unroll
    for (int i = 0; i < 4; ++i) {
      const int nt = wave * 4 + i;                         // 0..15
      const int n  = nt * 16 + m16;
      v8f acc = {};
      #pragma unroll
      for (int kc = 0; kc < 2; ++kc) {
        const int kb = kc * 32 + hi * 16;
        v8bf q0 = *(const v8bf*)&sWh[n * KH + kb];
        v8bf q1 = *(const v8bf*)&sWh[n * KH + kb + 8];
        acc = __builtin_amdgcn_wmma_f32_16x16x32_bf16(false, Af[kc],
                                                      false, cat8(q0, q1),
                                                      (short)0, acc, false, false);
      }
      #pragma unroll
      for (int r = 0; r < 8; ++r)
        sG[(r + 8 * hi) * NP + n] = acc[r];
    }
    __syncthreads();

    // ---- elementwise cell update: 16 rows x 50 units, gates from LDS ----
    const __bf16* gbuf = &sGx[t & 1][0];
    for (int u = tid; u < 16 * HID; u += 128) {
      const int m = u / HID, k = u - m * HID;
      const __bf16* gr = gbuf + m * NGP;
      const float gi = sG[m * NP + k]           + (float)gr[k];
      const float gj = sG[m * NP + HID + k]     + (float)gr[HID + k];
      const float gf = sG[m * NP + 2 * HID + k] + (float)gr[2 * HID + k];
      const float go = sG[m * NP + 3 * HID + k] + (float)gr[3 * HID + k];
      const float c  = sC[u];
      const float c2 = c * sigm(gf + 1.0f) + sigm(gi) * tanhf(gj);
      const float h2 = tanhf(c2) * sigm(go);
      sC[u] = c2;
      sH[m * KH + k] = (__bf16)h2;
      if (t == T_STEPS - 1) hfw[(size_t)(b0 + m) * HID + k] = h2;
    }
    wait_async0();          // next-step tile fully landed (this wave's share)
    __syncthreads();        // ... and everyone else's
  }
}

// ---------------------------------------------------------------------------
// Kernel 3: backward single cell (c0=h0=0) + BN + MLP head. 1 thread / row.
// ---------------------------------------------------------------------------
__global__ __launch_bounds__(128)
void head_kernel(const float* __restrict__ hfw, const __bf16* __restrict__ gbw,
                 const float* __restrict__ W1, const float* __restrict__ b1,
                 const float* __restrict__ W2, const float* __restrict__ b2,
                 float* __restrict__ out)
{
  const int b = blockIdx.x * 128 + threadIdx.x;
  if (b >= BATCH) return;
  const __bf16* g = gbw + (size_t)b * NGP;

  float feat[100];
  #pragma unroll
  for (int k = 0; k < HID; ++k) feat[k] = hfw[(size_t)b * HID + k] * BN_SCALE;
  #pragma unroll
  for (int k = 0; k < HID; ++k) {
    const float gi = (float)g[k], gj = (float)g[HID + k], go = (float)g[3 * HID + k];
    const float c2 = sigm(gi) * tanhf(gj);       // c0=0 => f-gate term vanishes
    feat[HID + k]  = tanhf(c2) * sigm(go) * BN_SCALE;
  }

  float o0 = b2[0], o1 = b2[1], o2 = b2[2];
  for (int j = 0; j < 25; ++j) {
    float s = b1[j];
    #pragma unroll
    for (int i = 0; i < 100; ++i) s += feat[i] * W1[i * 25 + j];
    s = s > 0.0f ? s : 0.0f;                     // relu
    o0 += s * W2[j * 3 + 0];
    o1 += s * W2[j * 3 + 1];
    o2 += s * W2[j * 3 + 2];
  }
  out[b * 3 + 0] = o0 * BN_SCALE;
  out[b * 3 + 1] = o1 * BN_SCALE;
  out[b * 3 + 2] = o2 * BN_SCALE;
}

// ---------------------------------------------------------------------------
extern "C" void kernel_launch(void* const* d_in, const int* in_sizes, int n_in,
                              void* d_out, int out_size, void* d_ws, size_t ws_size,
                              hipStream_t stream) {
  const int*   x1  = (const int*)  d_in[0];
  const int*   x2  = (const int*)  d_in[1];
  const float* emb = (const float*)d_in[2];
  const float* Wfw = (const float*)d_in[3];
  const float* bfw = (const float*)d_in[4];
  const float* Wbw = (const float*)d_in[5];
  const float* bbw = (const float*)d_in[6];
  const float* W1  = (const float*)d_in[7];
  const float* b1  = (const float*)d_in[8];
  const float* W2  = (const float*)d_in[9];
  const float* b2  = (const float*)d_in[10];
  float* out = (float*)d_out;

  char* ws = (char*)d_ws;
  size_t off = 0;
  __bf16* gxfw = (__bf16*)(ws + off); off += (size_t)T_STEPS * BATCH * NGP * 2;  // 109 MB
  __bf16* gxbw = (__bf16*)(ws + off); off += (size_t)BATCH * NGP * 2;
  float*  hfw  = (float*) (ws + off); off += (size_t)BATCH * HID * 4; off = (off + 255) & ~(size_t)255;
  __bf16* embbf  = (__bf16*)(ws + off); off += (size_t)VOCAB * EMBP * 2;         // 32 MB
  __bf16* wfragF = (__bf16*)(ws + off); off += (size_t)WFRAG_ELEMS * 2;
  __bf16* wfragB = (__bf16*)(ws + off); off += (size_t)WFRAG_ELEMS * 2;

  // 0) one-time precision/layout conversion passes (launch-invariant data)
  conv_emb_kernel  <<<(VOCAB * EMBP) / 256, 256, 0, stream>>>(emb, embbf);
  conv_wfrag_kernel<<<WFRAG_ELEMS   / 256, 256, 0, stream>>>(Wfw, wfragF);
  conv_wfrag_kernel<<<WFRAG_ELEMS   / 256, 256, 0, stream>>>(Wbw, wfragB);

  // 1) all-timestep input gates for the forward LSTM
  gates_kernel<<<dim3(T_STEPS, BATCH / 16), 128, 0, stream>>>(x1, x2, embbf, wfragF, bfw, gxfw, 0);
  // 2) backward LSTM needs only the last token's gates (h_bw[0] trick)
  gates_kernel<<<dim3(1, BATCH / 16), 128, 0, stream>>>(x1, x2, embbf, wfragB, bbw, gxbw, T_STEPS - 1);
  // 3) 256-step sequential scan, 64 batch tiles in parallel
  lstm_scan_kernel<<<BATCH / 16, 128, 0, stream>>>(gxfw, Wfw, hfw);
  // 4) backward cell + BN + MLP head
  head_kernel<<<BATCH / 128, 128, 0, stream>>>(hfw, gxbw, W1, b1, W2, b2, out);
}
